// TopKSAE_48816598287062
// MI455X (gfx1250) — matrix operand order repaired and checked by reading, not compiled, not run on hardware
//
#include <hip/hip_runtime.h>

typedef __attribute__((ext_vector_type(16))) _Float16 v16h;
typedef __attribute__((ext_vector_type(8)))  float    v8f;

#define D_IN   2048
#define D_HID  16384
#define NBATCH 8192
#define TOPK   64
#define KAUX   512

#define BM 256
#define BN 256
#define BK 32
#define NKSTEP (D_IN / BK)   // 64

#define USE_ASYNC_LDS 1

// ---- async global->LDS staging helpers (inline asm; sync fallback) --------
__device__ __forceinline__ void cp_b128(void* ldsp, const void* g) {
#if USE_ASYNC_LDS
  asm volatile("global_load_async_to_lds_b128 %0, %1, off"
               :
               : "v"((unsigned)(unsigned long long)ldsp),
                 "v"((unsigned long long)g)
               : "memory");
#else
  *(uint4*)ldsp = *(const uint4*)g;
#endif
}
__device__ __forceinline__ void cp_wait4() {
#if USE_ASYNC_LDS
  asm volatile("s_wait_asynccnt 0x4" ::: "memory");
#endif
}
__device__ __forceinline__ void cp_wait0() {
#if USE_ASYNC_LDS
  asm volatile("s_wait_asynccnt 0x0" ::: "memory");
#endif
}

// ---------------------------------------------------------------------------
// 1) x16 = f16(x - b_dec)
// ---------------------------------------------------------------------------
__global__ __launch_bounds__(256) void k_cvt_x(const float* __restrict__ x,
                                               const float* __restrict__ b_dec,
                                               _Float16* __restrict__ x16) {
  size_t i4   = (size_t)blockIdx.x * 256 + threadIdx.x;
  size_t base = i4 * 4;
  float4 xv = *(const float4*)(x + base);
  int col   = (int)(base & (D_IN - 1));
  float4 bv = *(const float4*)(b_dec + col);
  union { _Float16 h[4]; uint2 u2; } o;
  o.h[0] = (_Float16)(xv.x - bv.x);
  o.h[1] = (_Float16)(xv.y - bv.y);
  o.h[2] = (_Float16)(xv.z - bv.z);
  o.h[3] = (_Float16)(xv.w - bv.w);
  *(uint2*)(x16 + base) = o.u2;
}

// ---------------------------------------------------------------------------
// 2) w16 = f16(W_enc), same [D_HID][D_IN] layout
// ---------------------------------------------------------------------------
__global__ __launch_bounds__(256) void k_cvt_w(const float* __restrict__ w,
                                               _Float16* __restrict__ w16) {
  size_t i4   = (size_t)blockIdx.x * 256 + threadIdx.x;
  size_t base = i4 * 4;
  float4 wv = *(const float4*)(w + base);
  union { _Float16 h[4]; uint2 u2; } o;
  o.h[0] = (_Float16)wv.x; o.h[1] = (_Float16)wv.y;
  o.h[2] = (_Float16)wv.z; o.h[3] = (_Float16)wv.w;
  *(uint2*)(w16 + base) = o.u2;
}

// ---------------------------------------------------------------------------
// 3) WdT16[i][j] = f16(W_dec[j][i])   (W_dec: [D_IN][D_HID])
// ---------------------------------------------------------------------------
__global__ __launch_bounds__(256) void k_transpose(const float* __restrict__ Wd,
                                                   _Float16* __restrict__ WdT) {
  __shared__ float t[32][33];
  const int i0 = blockIdx.x * 32;  // hidden dim
  const int j0 = blockIdx.y * 32;  // input dim
  const int tx = threadIdx.x, ty = threadIdx.y;
#pragma unroll
  for (int k = 0; k < 4; ++k)
    t[ty + 8 * k][tx] = Wd[(size_t)(j0 + ty + 8 * k) * D_HID + i0 + tx];
  __syncthreads();
#pragma unroll
  for (int k = 0; k < 4; ++k)
    WdT[(size_t)(i0 + ty + 8 * k) * D_IN + j0 + tx] = (_Float16)t[tx][ty + 8 * k];
}

// ---------------------------------------------------------------------------
// 4) Encoder GEMM: C[m][n] = sum_k A16[m][k] * B16[n][k] + b_enc[n]
//    256x256 macro tile, BK=32, 16 waves (4x4), each wave 64x64 (4x4 WMMA),
//    async global->LDS double buffering.
//    LDS halves layout: A0 @0, B0 @10240, A1 @20480, B1 @30720  (81920 B)
// ---------------------------------------------------------------------------
__global__ __launch_bounds__(512) void k_enc_gemm(
    const _Float16* __restrict__ A,   // [NBATCH][D_IN]
    const _Float16* __restrict__ B,   // [D_HID][D_IN]
    const float* __restrict__ b_enc,
    float* __restrict__ C) {          // [NBATCH][D_HID]  (pre_acts)
  extern __shared__ _Float16 lds[];
  const int tid  = threadIdx.x;
  const int lane = tid & 31;
  const int wave = tid >> 5;                // 0..15
  const int wm   = wave >> 2;               // 0..3
  const int wn   = wave & 3;                // 0..3
  const int m0   = blockIdx.x * BM;
  const int n0   = blockIdx.y * BN;
  const int lrow = tid >> 1;                // 0..255 (load row)
  const int lcol = (tid & 1) * 16;          // half-offset 0 / 16
  const int fm   = lane & 15;
  const int ka   = (lane >> 4) * 8;         // A k-chunk base (ISA 16-bit A 16x32)
  const int kb   = (lane >> 4) * 8;         // B k-chunk base (same layout as A)

  v8f zero = {0.f, 0.f, 0.f, 0.f, 0.f, 0.f, 0.f, 0.f};
  v8f acc[4][4];
#pragma unroll
  for (int i = 0; i < 4; ++i)
#pragma unroll
    for (int j = 0; j < 4; ++j) acc[i][j] = zero;

  const _Float16* gA = A + (size_t)(m0 + lrow) * D_IN + lcol;
  const _Float16* gB = B + (size_t)(n0 + lrow) * D_IN + lcol;
  const int sOff = lrow * 40 + lcol;

  auto stage = [&](int kt) {
    const int s = kt & 1;
    _Float16* pA = lds + (s ? 20480 : 0) + sOff;
    _Float16* pB = lds + (s ? 30720 : 10240) + sOff;
    const _Float16* ga = gA + kt * BK;
    const _Float16* gb = gB + kt * BK;
    cp_b128(pA,     ga);
    cp_b128(pA + 8, ga + 8);
    cp_b128(pB,     gb);
    cp_b128(pB + 8, gb + 8);
  };

  stage(0);
  stage(1);

  for (int kt = 0; kt < NKSTEP; ++kt) {
    if (kt + 1 < NKSTEP) cp_wait4(); else cp_wait0();
    __syncthreads();
    const int s = kt & 1;
    const _Float16* lA = lds + (s ? 20480 : 0);
    const _Float16* lB = lds + (s ? 30720 : 10240);

    union F { v16h v; uint4 q[2]; };
    F af[4], bf[4];
#pragma unroll
    for (int i = 0; i < 4; ++i) {
      const int m = wm * 64 + i * 16 + fm;
      af[i].q[0] = *(const uint4*)&lA[m * 40 + ka];
      af[i].q[1] = *(const uint4*)&lA[m * 40 + ka + 16];
    }
#pragma unroll
    for (int j = 0; j < 4; ++j) {
      const int n = wn * 64 + j * 16 + fm;
      bf[j].q[0] = *(const uint4*)&lB[n * 40 + kb];
      bf[j].q[1] = *(const uint4*)&lB[n * 40 + kb + 16];
    }
#pragma unroll
    for (int i = 0; i < 4; ++i)
#pragma unroll
      for (int j = 0; j < 4; ++j)
        acc[i][j] = __builtin_amdgcn_wmma_f32_16x16x32_f16(
            false, af[i].v, false, bf[j].v, (short)0, acc[i][j], false, false);

    __syncthreads();
    if (kt + 2 < NKSTEP) stage(kt + 2);
  }

  const int mb0 = m0 + wm * 64 + (lane >> 4) * 8;
#pragma unroll
  for (int i = 0; i < 4; ++i) {
#pragma unroll
    for (int j = 0; j < 4; ++j) {
      const int n  = n0 + wn * 64 + j * 16 + fm;
      const float be = b_enc[n];
      float* cp = C + (size_t)(mb0 + i * 16) * D_HID + n;
#pragma unroll
      for (int r = 0; r < 8; ++r)
        cp[(size_t)r * D_HID] = acc[i][j][r] + be;
    }
  }
}

// ---------------------------------------------------------------------------
// 5) Per-row: exact radix select (top-64 global, top-512 among dead),
//    relu-scatter z in place, deterministic index-ordered compaction.
// ---------------------------------------------------------------------------
__global__ __launch_bounds__(512) void k_topk(
    float* __restrict__ z,                         // in: pre_acts, out: z
    const unsigned char* __restrict__ dead_mask,   // [D_HID]
    int* __restrict__ top_idx, float* __restrict__ top_val,
    int* __restrict__ aux_idx, float* __restrict__ aux_val) {
  extern __shared__ unsigned int smem[];
  unsigned int* u     = smem;           // D_HID monotonic keys
  unsigned int* dbits = smem + D_HID;   // 512 words of dead bits
  __shared__ unsigned int s_cnt;
  __shared__ unsigned int s_scan[512];

  const int tid = threadIdx.x;
  const int row = blockIdx.x;
  float* zr = z + (size_t)row * D_HID;

  for (int i = tid; i < D_HID; i += 512) {
    unsigned int b = __float_as_uint(zr[i]);
    u[i] = (b & 0x80000000u) ? ~b : (b | 0x80000000u);
  }
  {
    unsigned int w = 0;
    const unsigned char* dm = dead_mask + tid * 32;
#pragma unroll
    for (int b = 0; b < 32; ++b) w |= (dm[b] ? 1u : 0u) << b;
    dbits[tid] = w;
  }
  __syncthreads();

  // ---- radix select: 64th largest over all features
  unsigned int th1;
  {
    unsigned int prefix = 0;
    int k = TOPK;
    for (int bit = 31; bit >= 0; --bit) {
      if (tid == 0) s_cnt = 0;
      __syncthreads();
      const unsigned int target = (prefix >> bit) | 1u;
      int local = 0;
      for (int i = tid; i < D_HID; i += 512) local += ((u[i] >> bit) == target);
      if (local) atomicAdd(&s_cnt, (unsigned int)local);
      __syncthreads();
      const unsigned int c = s_cnt;
      __syncthreads();
      if ((int)c >= k) prefix |= (1u << bit); else k -= (int)c;
    }
    th1 = prefix;
  }
  // ---- radix select: 512th largest among dead features
  unsigned int th2;
  {
    unsigned int prefix = 0;
    int k = KAUX;
    for (int bit = 31; bit >= 0; --bit) {
      if (tid == 0) s_cnt = 0;
      __syncthreads();
      const unsigned int target = (prefix >> bit) | 1u;
      int local = 0;
      for (int i = tid; i < D_HID; i += 512)
        if ((dbits[i >> 5] >> (i & 31)) & 1u) local += ((u[i] >> bit) == target);
      if (local) atomicAdd(&s_cnt, (unsigned int)local);
      __syncthreads();
      const unsigned int c = s_cnt;
      __syncthreads();
      if ((int)c >= k) prefix |= (1u << bit); else k -= (int)c;
    }
    th2 = prefix;
  }

  // ---- write dense z (strided, coalesced)
  for (int i = tid; i < D_HID; i += 512) {
    const unsigned int ui = u[i];
    const unsigned int ob = (ui & 0x80000000u) ? (ui & 0x7FFFFFFFu) : ~ui;
    const float v = __uint_as_float(ob);
    zr[i] = (ui >= th1) ? fmaxf(v, 0.0f) : 0.0f;
  }

  // ---- deterministic compaction: each thread owns 32 consecutive features
  const int i0 = tid * 32;
  int c1 = 0, c2 = 0;
  const unsigned int myd = dbits[tid];
#pragma unroll
  for (int t = 0; t < 32; ++t) {
    const unsigned int ui = u[i0 + t];
    c1 += (ui >= th1);
    c2 += (((myd >> t) & 1u) && ui >= th2);
  }
  s_scan[tid] = (unsigned int)c1;
  __syncthreads();
  for (int off = 1; off < 512; off <<= 1) {
    unsigned int v = s_scan[tid];
    unsigned int a = (tid >= off) ? s_scan[tid - off] : 0u;
    __syncthreads();
    s_scan[tid] = v + a;
    __syncthreads();
  }
  int pos1 = (int)s_scan[tid] - c1;
  const int tot1 = (int)s_scan[511];
  __syncthreads();
  s_scan[tid] = (unsigned int)c2;
  __syncthreads();
  for (int off = 1; off < 512; off <<= 1) {
    unsigned int v = s_scan[tid];
    unsigned int a = (tid >= off) ? s_scan[tid - off] : 0u;
    __syncthreads();
    s_scan[tid] = v + a;
    __syncthreads();
  }
  int pos2 = (int)s_scan[tid] - c2;
  const int tot2 = (int)s_scan[511];

  int*   ti = top_idx + (size_t)row * TOPK;
  float* tv = top_val + (size_t)row * TOPK;
  int*   ai = aux_idx + (size_t)row * KAUX;
  float* av = aux_val + (size_t)row * KAUX;
#pragma unroll
  for (int t = 0; t < 32; ++t) {
    const int i = i0 + t;
    const unsigned int ui = u[i];
    const unsigned int ob = (ui & 0x80000000u) ? (ui & 0x7FFFFFFFu) : ~ui;
    const float rv = fmaxf(__uint_as_float(ob), 0.0f);
    if (ui >= th1) { if (pos1 < TOPK) { ti[pos1] = i; tv[pos1] = rv; } ++pos1; }
    if (((myd >> t) & 1u) && ui >= th2) {
      if (pos2 < KAUX) { ai[pos2] = i; av[pos2] = rv; } ++pos2;
    }
  }
  // pad (harmless zeros)
  for (int p = (tot1 < TOPK ? tot1 : TOPK) + tid; p < TOPK; p += 512) { ti[p] = 0; tv[p] = 0.0f; }
  for (int p = (tot2 < KAUX ? tot2 : KAUX) + tid; p < KAUX; p += 512) { ai[p] = 0; av[p] = 0.0f; }
}

// ---------------------------------------------------------------------------
// 6) Sparse decode: x_hat = sum_t val*WdT[idx] + b_dec ; e = x - x_hat ;
//    e_hat = sum_t aux_val*WdT[aux_idx]; per-row loss partials (deterministic).
// ---------------------------------------------------------------------------
__global__ __launch_bounds__(256) void k_decode(
    const float* __restrict__ x, const float* __restrict__ b_dec,
    const _Float16* __restrict__ WdT,  // [D_HID][D_IN], lives in L2
    const int* __restrict__ top_idx, const float* __restrict__ top_val,
    const int* __restrict__ aux_idx, const float* __restrict__ aux_val,
    float* __restrict__ x_hat, float* __restrict__ rowstats) {
  const int row = blockIdx.x, tid = threadIdx.x;
  const int j0 = tid * 8;
  float xv[8], acc[8];
  {
    float4 a = *(const float4*)(x + (size_t)row * D_IN + j0);
    float4 b = *(const float4*)(x + (size_t)row * D_IN + j0 + 4);
    xv[0] = a.x; xv[1] = a.y; xv[2] = a.z; xv[3] = a.w;
    xv[4] = b.x; xv[5] = b.y; xv[6] = b.z; xv[7] = b.w;
    float4 c = *(const float4*)(b_dec + j0);
    float4 d = *(const float4*)(b_dec + j0 + 4);
    acc[0] = c.x; acc[1] = c.y; acc[2] = c.z; acc[3] = c.w;
    acc[4] = d.x; acc[5] = d.y; acc[6] = d.z; acc[7] = d.w;
  }
  const int*   ti = top_idx + (size_t)row * TOPK;
  const float* tv = top_val + (size_t)row * TOPK;
  float l0c = 0.0f;
  for (int t = 0; t < TOPK; ++t) {
    const float val = tv[t];
    const int   idx = ti[t];
    if (tid == 0) l0c += (val > 0.0f) ? 1.0f : 0.0f;
    union { uint4 q; _Float16 h[8]; } w;
    w.q = *(const uint4*)(WdT + (size_t)idx * D_IN + j0);
#pragma unroll
    for (int c = 0; c < 8; ++c) acc[c] = fmaf(val, (float)w.h[c], acc[c]);
  }
  float recon = 0.0f, e[8];
#pragma unroll
  for (int c = 0; c < 8; ++c) { float d = acc[c] - xv[c]; recon += d * d; e[c] = -d; }
  {
    float4 o0 = { acc[0], acc[1], acc[2], acc[3] };
    float4 o1 = { acc[4], acc[5], acc[6], acc[7] };
    *(float4*)(x_hat + (size_t)row * D_IN + j0)     = o0;
    *(float4*)(x_hat + (size_t)row * D_IN + j0 + 4) = o1;
  }
  const int*   ai = aux_idx + (size_t)row * KAUX;
  const float* av = aux_val + (size_t)row * KAUX;
  float acc2[8] = {0, 0, 0, 0, 0, 0, 0, 0};
  for (int t = 0; t < KAUX; ++t) {
    const float val = av[t];
    const int   idx = ai[t];
    if (t + 8 < KAUX)
      __builtin_prefetch(WdT + (size_t)ai[t + 8] * D_IN + j0, 0, 1);
    union { uint4 q; _Float16 h[8]; } w;
    w.q = *(const uint4*)(WdT + (size_t)idx * D_IN + j0);
#pragma unroll
    for (int c = 0; c < 8; ++c) acc2[c] = fmaf(val, (float)w.h[c], acc2[c]);
  }
  float auxsq = 0.0f;
#pragma unroll
  for (int c = 0; c < 8; ++c) { float d = acc2[c] - e[c]; auxsq += d * d; }

  __shared__ float r1[256], r2[256];
  r1[tid] = recon; r2[tid] = auxsq;
  __syncthreads();
  for (int off = 128; off > 0; off >>= 1) {
    if (tid < off) { r1[tid] += r1[tid + off]; r2[tid] += r2[tid + off]; }
    __syncthreads();
  }
  if (tid == 0) {
    rowstats[row * 4 + 0] = r1[0];
    rowstats[row * 4 + 1] = r2[0];
    rowstats[row * 4 + 2] = l0c;
  }
}

// ---------------------------------------------------------------------------
// 7) Final fixed-order reduction -> loss, recon, aux, l0
// ---------------------------------------------------------------------------
__global__ __launch_bounds__(512) void k_reduce(const float* __restrict__ rowstats,
                                                float* __restrict__ scal) {
  __shared__ float r1[512], r2[512], r3[512];
  const int tid = threadIdx.x;
  float a = 0.f, b = 0.f, c = 0.f;
  for (int r = tid; r < NBATCH; r += 512) {
    a += rowstats[r * 4 + 0];
    b += rowstats[r * 4 + 1];
    c += rowstats[r * 4 + 2];
  }
  r1[tid] = a; r2[tid] = b; r3[tid] = c;
  __syncthreads();
  for (int off = 256; off > 0; off >>= 1) {
    if (tid < off) { r1[tid] += r1[tid + off]; r2[tid] += r2[tid + off]; r3[tid] += r3[tid + off]; }
    __syncthreads();
  }
  if (tid == 0) {
    const float inv = 1.0f / ((float)NBATCH * (float)D_IN);
    const float recon = r1[0] * inv;
    const float aux   = r2[0] * inv;
    const float l0    = r3[0] / (float)NBATCH;
    scal[0] = recon + aux * (1.0f / 32.0f);
    scal[1] = recon;
    scal[2] = aux;
    scal[3] = l0;
  }
}

// ---------------------------------------------------------------------------
extern "C" void kernel_launch(void* const* d_in, const int* in_sizes, int n_in,
                              void* d_out, int out_size, void* d_ws, size_t ws_size,
                              hipStream_t stream) {
  (void)in_sizes; (void)n_in; (void)out_size; (void)ws_size;
  const float* x     = (const float*)d_in[0];
  const float* W_enc = (const float*)d_in[1];
  const float* b_enc = (const float*)d_in[2];
  const float* W_dec = (const float*)d_in[3];
  const float* b_dec = (const float*)d_in[4];
  const unsigned char* dead = (const unsigned char*)d_in[5];

  char* ws = (char*)d_ws;
  _Float16* X16     = (_Float16*)(ws);                 // 32 MB
  _Float16* We16    = (_Float16*)(ws + 33554432ull);   // 64 MB
  _Float16* WdT16   = (_Float16*)(ws + 100663296ull);  // 64 MB
  int*      top_idx = (int*)   (ws + 167772160ull);    // 2 MB
  float*    top_val = (float*) (ws + 169869312ull);    // 2 MB
  int*      aux_idx = (int*)   (ws + 171966464ull);    // 16 MB
  float*    aux_val = (float*) (ws + 188743680ull);    // 16 MB
  float*    rowstats= (float*) (ws + 205520896ull);    // 128 KB

  float* x_hat = (float*)d_out;
  float* z     = x_hat + (size_t)NBATCH * D_IN;
  float* scal  = z + (size_t)NBATCH * D_HID;

  k_cvt_x<<<(NBATCH * D_IN / 4) / 256, 256, 0, stream>>>(x, b_dec, X16);
  k_cvt_w<<<(D_HID * D_IN / 4) / 256, 256, 0, stream>>>(W_enc, We16);
  k_transpose<<<dim3(D_HID / 32, D_IN / 32), dim3(32, 8), 0, stream>>>(W_dec, WdT16);
  k_enc_gemm<<<dim3(NBATCH / BM, D_HID / BN), 512, 81920, stream>>>(X16, We16, b_enc, z);
  const size_t lds = (size_t)(D_HID + 512) * sizeof(unsigned int);
  k_topk<<<NBATCH, 512, lds, stream>>>(z, dead, top_idx, top_val, aux_idx, aux_val);
  k_decode<<<NBATCH, 256, 0, stream>>>(x, b_dec, WdT16, top_idx, top_val,
                                       aux_idx, aux_val, x_hat, rowstats);
  k_reduce<<<1, 512, 0, stream>>>(rowstats, scal);
}
